// Attention_49658411876662
// MI455X (gfx1250) — matrix operand order repaired and checked by reading, not compiled, not run on hardware
//
#include <hip/hip_runtime.h>

// MI455X / gfx1250, wave32. All three GEMMs run on v_wmma_f32_16x16x32_f16.
// Shapes: B=256, N=196 (pad 208/224), Cd=2048, Qd=1024, E=1024, G=8 (pad 16).
// Stage-1 GEMM uses register-prefetch software pipelining: global loads for
// K-chunk k+1 are issued before the WMMA burst of chunk k, hiding HBM/L2
// latency behind the matrix pipe with a single LDS buffer (<64KB static LDS).

typedef __attribute__((ext_vector_type(16))) _Float16 v16h;
typedef __attribute__((ext_vector_type(8)))  _Float16 v8h;
typedef __attribute__((ext_vector_type(4)))  _Float16 v4h;
typedef __attribute__((ext_vector_type(8)))  float    v8f;
typedef __attribute__((ext_vector_type(4)))  float    v4f;

#define WMMA_F16(A, B, C) \
  __builtin_amdgcn_wmma_f32_16x16x32_f16(false, (A), false, (B), (short)0, (C), false, false)

// A-fragment: lane holds row M=l%16; 16 contiguous K values
__device__ __forceinline__ v16h ld_fragA(const _Float16* p) {
  v8h lo = *(const v8h*)p;
  v8h hi = *(const v8h*)(p + 8);
  v16h r;
#pragma unroll
  for (int i = 0; i < 8; ++i) { r[i] = lo[i]; r[8 + i] = hi[i]; }
  return r;
}
// B-fragment: lane holds col N=l%16; 16 contiguous K values
__device__ __forceinline__ v16h ld_fragB(const _Float16* p) {
  v8h lo = *(const v8h*)p;
  v8h hi = *(const v8h*)(p + 8);
  v16h r;
#pragma unroll
  for (int i = 0; i < 8; ++i) { r[i] = lo[i]; r[8 + i] = hi[i]; }
  return r;
}

// ---------------- Kernel 1: QP[b,e] = query @ Wq^T + bq  (256x1024, K=1024) ----
__global__ __launch_bounds__(256) void qp_kernel(const float* __restrict__ query,
                                                 const float* __restrict__ Wq,
                                                 const float* __restrict__ bq,
                                                 float* __restrict__ QP) {
  const int w = threadIdx.x >> 5, lane = threadIdx.x & 31;
  const int hl = lane >> 4, r = lane & 15;
  const int etile = blockIdx.x * 8 + w;   // 0..63
  const int btile = blockIdx.y;           // 0..15
  const float* qrow = query + (size_t)(btile * 16 + r) * 1024;
  const float* wrow = Wq + (size_t)(etile * 16 + r) * 1024;
  v8f acc = {0.f, 0.f, 0.f, 0.f, 0.f, 0.f, 0.f, 0.f};
  for (int k0 = 0; k0 < 1024; k0 += 32) {
    v16h af, bf;
    const float* qa = qrow + k0 + hl * 16;
#pragma unroll
    for (int i = 0; i < 8; ++i) { af[i] = (_Float16)qa[i]; af[8 + i] = (_Float16)qa[8 + i]; }
    const float* wb = wrow + k0 + hl * 16;
#pragma unroll
    for (int i = 0; i < 16; ++i) bf[i] = (_Float16)wb[i];
    acc = WMMA_F16(af, bf, acc);
  }
  const int e = etile * 16 + r;
  const float bqv = bq[e];
#pragma unroll
  for (int a = 0; a < 8; ++a)
    QP[(size_t)(btile * 16 + a + 8 * hl) * 1024 + e] = acc[a] + bqv;
}

// ---------------- Kernel 2: fused projection+tanh+attn+softmax+glimpse --------
// grid = 256 (one block per batch), block = 256 threads (8 waves)
__global__ __launch_bounds__(256) void attn_kernel(const float* __restrict__ context,
                                                   const float* __restrict__ Wc,
                                                   const float* __restrict__ bc,
                                                   const float* __restrict__ Wo,
                                                   const float* __restrict__ QP,
                                                   float* __restrict__ out) {
  // sA: context chunk 208 x (64 k) f16, pitch 72  -> 29952 B
  // sB: Wc chunk     128 x (64 k) f16, pitch 72   -> 18432 B
  // sC: combined (alias of sA+sB) 112 x 128 f16, pitch 136 -> 30464 B
  __shared__ __align__(16) char smem[48384];
  __shared__ __align__(16) float sAttn[208 * 8];        // 6656 B
  __shared__ __align__(16) _Float16 sWT[16 * 232];      // 7424 B  (weights^T, padded)
  _Float16* sA = (_Float16*)smem;
  _Float16* sB = (_Float16*)(smem + 29952);
  _Float16* sC = (_Float16*)smem;

  const int b = blockIdx.x;
  const int w = threadIdx.x >> 5, lane = threadIdx.x & 31;
  const int hl = lane >> 4, r = lane & 15;
  const float* ctx = context + (size_t)b * (196 * 2048);
  const float* qpb = QP + (size_t)b * 1024;

  // per-thread staging coordinates (3328 = 13*256 A units, 2048 = 8*256 B units)
  const v8f z8 = {0.f, 0.f, 0.f, 0.f, 0.f, 0.f, 0.f, 0.f};
  v8f attA = z8, attB = z8;   // wave w's attn tiles for row-tiles w and 7+w

  // ---- Stage 1: combined = tanh(context@Wc^T + qp + bc), attn += combined@Wo^T
  for (int ec = 0; ec < 8; ++ec) {
    const int e0 = ec * 128;
    const float addv = qpb[e0 + w * 16 + r] + bc[e0 + w * 16 + r];
    v8f acc[13];
#pragma unroll
    for (int i = 0; i < 13; ++i) acc[i] = z8;

    v4f pa[13], pb[8];
    // prefetch K-chunk 0 into registers
#pragma unroll
    for (int s2 = 0; s2 < 13; ++s2) {
      const int t = (int)threadIdx.x + s2 * 256;
      const int row = t >> 4, cq = (t & 15) << 2;
      v4f x = {0.f, 0.f, 0.f, 0.f};
      if (row < 196) x = *(const v4f*)(ctx + (size_t)row * 2048 + cq);
      pa[s2] = x;
    }
#pragma unroll
    for (int s2 = 0; s2 < 8; ++s2) {
      const int t = (int)threadIdx.x + s2 * 256;
      const int row = t >> 4, cq = (t & 15) << 2;
      pb[s2] = *(const v4f*)(Wc + (size_t)(e0 + row) * 2048 + cq);
    }

    for (int kc = 0; kc < 32; ++kc) {     // K = 2048 in chunks of 64
      const int k0n = (kc + 1) * 64;
      __syncthreads();                    // all readers of previous LDS chunk done
      // write prefetched chunk to LDS (f32 -> f16)
#pragma unroll
      for (int s2 = 0; s2 < 13; ++s2) {
        const int t = (int)threadIdx.x + s2 * 256;
        const int row = t >> 4, cq = (t & 15) << 2;
        v4h h4;
#pragma unroll
        for (int i = 0; i < 4; ++i) h4[i] = (_Float16)pa[s2][i];
        *(v4h*)(sA + row * 72 + cq) = h4;
      }
#pragma unroll
      for (int s2 = 0; s2 < 8; ++s2) {
        const int t = (int)threadIdx.x + s2 * 256;
        const int row = t >> 4, cq = (t & 15) << 2;
        v4h h4;
#pragma unroll
        for (int i = 0; i < 4; ++i) h4[i] = (_Float16)pb[s2][i];
        *(v4h*)(sB + row * 72 + cq) = h4;
      }
      // issue next chunk's global loads; they fly during the WMMA burst below
      if (kc < 31) {
#pragma unroll
        for (int s2 = 0; s2 < 13; ++s2) {
          const int t = (int)threadIdx.x + s2 * 256;
          const int row = t >> 4, cq = (t & 15) << 2;
          v4f x = {0.f, 0.f, 0.f, 0.f};
          if (row < 196) x = *(const v4f*)(ctx + (size_t)row * 2048 + k0n + cq);
          pa[s2] = x;
        }
#pragma unroll
        for (int s2 = 0; s2 < 8; ++s2) {
          const int t = (int)threadIdx.x + s2 * 256;
          const int row = t >> 4, cq = (t & 15) << 2;
          pb[s2] = *(const v4f*)(Wc + (size_t)(e0 + row) * 2048 + k0n + cq);
        }
      }
      __syncthreads();
#pragma unroll
      for (int s = 0; s < 2; ++s) {
        const int ks = s * 32;
        v16h bf = ld_fragB(sB + (w * 16 + r) * 72 + ks + hl * 16);
#pragma unroll
        for (int rt = 0; rt < 13; ++rt) {
          v16h af = ld_fragA(sA + (rt * 16 + r) * 72 + ks + hl * 16);
          acc[rt] = WMMA_F16(af, bf, acc[rt]);
        }
      }
    }
    // fuse bias + tanh in registers
#pragma unroll
    for (int rt = 0; rt < 13; ++rt)
#pragma unroll
      for (int a = 0; a < 8; ++a) acc[rt][a] = tanhf(acc[rt][a] + addv);

    // attn accumulation in two row-halves, staging combined as f16 through LDS
#pragma unroll
    for (int hh = 0; hh < 2; ++hh) {
      const int rtBeg = hh ? 7 : 0;
      const int cnt = hh ? 6 : 7;
      __syncthreads();   // previous readers of sA/sB/sC done
#pragma unroll
      for (int j = 0; j < 7; ++j) {
        if (j < cnt) {
#pragma unroll
          for (int a = 0; a < 8; ++a)
            sC[(j * 16 + a + 8 * hl) * 136 + w * 16 + r] = (_Float16)acc[rtBeg + j][a];
        }
      }
      __syncthreads();
      if (w < cnt) {                  // wave-uniform: EXEC all ones inside
        v8f& att = hh ? attB : attA;
#pragma unroll
        for (int kk = 0; kk < 4; ++kk) {
          v16h af = ld_fragA(sC + (w * 16 + r) * 136 + kk * 32 + hl * 16);
          v16h bf;
#pragma unroll
          for (int i = 0; i < 16; ++i) bf[i] = (_Float16)0.f;
          if (r < 8) {                // g columns 8..15 are zero padding
            const float* wo = Wo + (size_t)r * 1024 + e0 + kk * 32 + hl * 16;
#pragma unroll
            for (int i = 0; i < 16; ++i) bf[i] = (_Float16)wo[i];
          }
          att = WMMA_F16(af, bf, att);
        }
      }
    }
  }

  // dump attn tiles (waves own disjoint row-tiles, no atomics needed)
  if (r < 8) {
#pragma unroll
    for (int a = 0; a < 8; ++a) {
      if (w < 7) sAttn[(w * 16 + a + 8 * hl) * 8 + r] = attA[a];
      if (w < 6) sAttn[((7 + w) * 16 + a + 8 * hl) * 8 + r] = attB[a];
    }
  }
  __syncthreads();
  // zero padded weights^T buffer (rows g>=8 and cols n>=196 must be 0)
  for (int t = threadIdx.x; t < 928; t += 256) ((unsigned long long*)sWT)[t] = 0ull;
  __syncthreads();

  // ---- Stage 2: softmax over n (196) per g; bo cancels in softmax ----
  {
    const int g = w;                  // 8 waves <-> 8 glimpse columns
    float v[7];
    float m = -3.0e38f;
#pragma unroll
    for (int i = 0; i < 7; ++i) {
      const int n = lane + 32 * i;
      v[i] = (n < 196) ? sAttn[n * 8 + g] : -3.0e38f;
      m = fmaxf(m, v[i]);
    }
#pragma unroll
    for (int off = 16; off > 0; off >>= 1) m = fmaxf(m, __shfl_xor(m, off, 32));
    float s = 0.f;
#pragma unroll
    for (int i = 0; i < 7; ++i) {
      const int n = lane + 32 * i;
      const float e = (n < 196) ? __expf(v[i] - m) : 0.f;
      v[i] = e;
      s += e;
    }
#pragma unroll
    for (int off = 16; off > 0; off >>= 1) s += __shfl_xor(s, off, 32);
    const float inv = 1.f / s;
#pragma unroll
    for (int i = 0; i < 7; ++i) {
      const int n = lane + 32 * i;
      if (n < 196) sWT[g * 232 + n] = (_Float16)(v[i] * inv);
    }
  }
  __syncthreads();

  // ---- Stage 3: glimpses[g, c] = weights^T @ context  (M=16pad, K=224pad) ----
#pragma unroll 1
  for (int cc = 0; cc < 8; ++cc) {
    const int c0 = cc * 256;
    v8f g0 = z8, g1 = z8;
#pragma unroll 1
    for (int kk = 0; kk < 7; ++kk) {
      v16h af = ld_fragA(sWT + r * 232 + kk * 32 + hl * 16);
#pragma unroll
      for (int t = 0; t < 2; ++t) {
        const int c = c0 + (w * 2 + t) * 16 + r;
        v16h bf;
#pragma unroll
        for (int vv = 0; vv < 8; ++vv) {
          const int n = kk * 32 + hl * 16 + 2 * vv;
          const float x0 = (n < 196) ? ctx[(size_t)n * 2048 + c] : 0.f;
          const float x1 = (n + 1 < 196) ? ctx[(size_t)(n + 1) * 2048 + c] : 0.f;
          bf[2 * vv] = (_Float16)x0;
          bf[2 * vv + 1] = (_Float16)x1;
        }
        if (t == 0) g0 = WMMA_F16(af, bf, g0);
        else        g1 = WMMA_F16(af, bf, g1);
      }
    }
    if (hl == 0) {                    // rows m=0..7 <-> g
#pragma unroll
      for (int a = 0; a < 8; ++a) {
        out[(size_t)b * 16384 + (size_t)a * 2048 + c0 + (w * 2 + 0) * 16 + r] = g0[a];
        out[(size_t)b * 16384 + (size_t)a * 2048 + c0 + (w * 2 + 1) * 16 + r] = g1[a];
      }
    }
  }
}

extern "C" void kernel_launch(void* const* d_in, const int* in_sizes, int n_in,
                              void* d_out, int out_size, void* d_ws, size_t ws_size,
                              hipStream_t stream) {
  (void)in_sizes; (void)n_in; (void)out_size; (void)ws_size;
  const float* context = (const float*)d_in[0];
  const float* query   = (const float*)d_in[1];
  const float* Wq      = (const float*)d_in[2];
  const float* bq      = (const float*)d_in[3];
  const float* Wc      = (const float*)d_in[4];
  const float* bc      = (const float*)d_in[5];
  const float* Wo      = (const float*)d_in[6];
  // bo (d_in[7]) is a per-g constant along the softmax axis -> cancels exactly.
  float* out = (float*)d_out;
  float* QP  = (float*)d_ws;  // 256*1024 floats = 1 MB scratch

  qp_kernel<<<dim3(8, 16), 256, 0, stream>>>(query, Wq, bq, QP);
  attn_kernel<<<dim3(256), 256, 0, stream>>>(context, Wc, bc, Wo, QP, out);
}
